// TransformerBlock_4681514353259
// MI455X (gfx1250) — matrix operand order repaired and checked by reading, not compiled, not run on hardware
//
#include <hip/hip_runtime.h>
#include <hip/hip_bf16.h>
#include <math.h>
#include <stdint.h>

// ---------------------------------------------------------------------------
// Transformer block for MI455X (gfx1250, wave32, WMMA + Tensor Data Mover).
// bf16 WMMA 16x16x32 with fp32 accumulation for all matmuls.
// ---------------------------------------------------------------------------

typedef unsigned short u16;
typedef __bf16 bf16_t;
typedef __attribute__((ext_vector_type(16))) bf16_t v16bf;
typedef __attribute__((ext_vector_type(8)))  float  v8f;
typedef __attribute__((ext_vector_type(4)))  unsigned int v4u;
typedef __attribute__((ext_vector_type(8)))  int v8i;
typedef __attribute__((ext_vector_type(4)))  int v4i;

#define D_MODEL 1024
#define T_SEQ   2048
#define BATCH   4
#define NHEADS  16
#define DHEAD   64
#define DFF     4096
#define RMS_EPS 1e-5f

#if __has_builtin(__builtin_amdgcn_tensor_load_to_lds) && \
    __has_builtin(__builtin_amdgcn_s_wait_tensorcnt)
#define HAS_TDM 1
#else
#define HAS_TDM 0
#endif

__device__ __forceinline__ u16 f2bf(float f) {
  unsigned u = __float_as_uint(f);
  u += 0x7FFFu + ((u >> 16) & 1u);          // round-to-nearest-even
  return (u16)(u >> 16);
}

union Frag16 {        // one WMMA A/B operand: 16 bf16 = 32 bytes = 8 VGPRs
  uint4 q[2];
  u16   s[16];
  v16bf b;
};
union Frag32 {        // 32 bf16 staging for transposed LDS scatter
  uint4 q[4];
  u16   s[32];
};

// LDS byte offset (within workgroup allocation) of a __shared__ object:
// generic -> addrspace(3) cast, then ptrtoint.
typedef __attribute__((address_space(3))) char lds_char;
__device__ __forceinline__ unsigned lds_addr_of(void* p) {
  return (unsigned)(uintptr_t)(lds_char*)p;
}

#if HAS_TDM
// 2D tensor_load_to_lds: tile (tile_d0 x tile_d1) of u16 elements, row stride
// stride_d0 (elements), optional LDS row padding via pad codes
// (interval code c -> 2^(c+1) DWORDs, amount code a -> (a+1) DWORDs).
__device__ __forceinline__ void tdm_load_2d_u16(
    unsigned lds_byte_addr, const void* gptr,
    unsigned tile_d0, unsigned tile_d1,
    unsigned tensor_d0, unsigned tensor_d1,
    unsigned long long stride_d0,
    unsigned pad_interval_code, unsigned pad_amount_code) {
  unsigned long long ga = (unsigned long long)(uintptr_t)gptr;
  v4u g0;
  g0.x = 1u;                                     // count=1, user mode
  g0.y = lds_byte_addr;                          // D#.lds_addr
  g0.z = (unsigned)(ga & 0xFFFFFFFFu);           // global_addr[31:0]
  g0.w = (unsigned)((ga >> 32) & 0x01FFFFFFu)    // global_addr[56:32]
       | (2u << 30);                             // type=2 ("image")
  v8i g1;
  g1[0] = (int)((1u << 16)                       // data_size = 2 bytes
              | (1u << 20)                       // pad_enable
              | (pad_interval_code << 22)
              | (pad_amount_code << 25));
  g1[1] = (int)((tensor_d0 & 0xFFFFu) << 16);    // abar_addr=0 | td0[15:0]
  g1[2] = (int)(((tensor_d0 >> 16) & 0xFFFFu) | ((tensor_d1 & 0xFFFFu) << 16));
  g1[3] = (int)(((tensor_d1 >> 16) & 0xFFFFu) | ((tile_d0 & 0xFFFFu) << 16));
  g1[4] = (int)(tile_d1 & 0xFFFFu);              // tile_dim1 | tile_dim2=0
  g1[5] = (int)(stride_d0 & 0xFFFFFFFFu);        // tensor_dim0_stride
  g1[6] = (int)((stride_d0 >> 32) & 0xFFFFu);    // | td1_stride=0 (2D)
  g1[7] = 0;
  v4i z4 = {0, 0, 0, 0};
#if defined(__clang_major__) && __clang_major__ >= 23
  v8i z8 = {0, 0, 0, 0, 0, 0, 0, 0};
  __builtin_amdgcn_tensor_load_to_lds(g0, g1, z4, z4, z8, 0);
#else
  __builtin_amdgcn_tensor_load_to_lds(g0, g1, z4, z4, 0);
#endif
}
#endif  // HAS_TDM

// ---------------------------------------------------------------------------
// fp32 -> bf16 conversion (weights)
// ---------------------------------------------------------------------------
__global__ void cvt_f32_bf16(const float* __restrict__ in,
                             u16* __restrict__ out, int n) {
  int i = blockIdx.x * blockDim.x + threadIdx.x;
  int stride = gridDim.x * blockDim.x;
  for (; i < n; i += stride) out[i] = f2bf(in[i]);
}

// ---------------------------------------------------------------------------
// RMSNorm: fp32 row in, bf16 normalized row out. One block per row (1024).
// ---------------------------------------------------------------------------
__global__ __launch_bounds__(256) void rmsnorm_bf16(
    const float* __restrict__ x, const float* __restrict__ g,
    u16* __restrict__ out) {
  const int row = blockIdx.x;
  const int tid = threadIdx.x;
  const float* xr = x + (size_t)row * D_MODEL;
  float v[4];
  float ss = 0.f;
#pragma unroll
  for (int i = 0; i < 4; ++i) { v[i] = xr[tid + i * 256]; ss += v[i] * v[i]; }
  __shared__ float red[256];
  red[tid] = ss;
  __syncthreads();
  for (int sft = 128; sft > 0; sft >>= 1) {
    if (tid < sft) red[tid] += red[tid + sft];
    __syncthreads();
  }
  const float rinv = rsqrtf(red[0] * (1.0f / D_MODEL) + RMS_EPS);
  u16* orow = out + (size_t)row * D_MODEL;
#pragma unroll
  for (int i = 0; i < 4; ++i) {
    const int c = tid + i * 256;
    orow[c] = f2bf(v[i] * rinv * g[c]);
  }
}

// ---------------------------------------------------------------------------
// bf16 GEMM, 128x128 tile, BK=32, 256 threads = 8 waves.
// Wave grid 4(M) x 2(N): each wave -> 32x64 = 2x4 WMMA accumulators.
// A tile [m][k] (stride 40 u16), B tile row-major [k][n] (stride 136 u16).
// TDM path: wave 0 DMAs stage s+1 into the other LDS buffer while all waves
// compute stage s; s_wait_tensorcnt 0 + barrier closes each stage.
// EPI: 0 = store bf16, 1 = exact GELU -> bf16, 2 = + residual -> fp32
// ---------------------------------------------------------------------------
template <int EPI>
__global__ __launch_bounds__(256) void gemm_bf16_128x128(
    const u16* __restrict__ A, const u16* __restrict__ B,
    void* __restrict__ Cout, const float* __restrict__ Res,
    int M, int N, int K) {
  __shared__ __align__(16) u16 As[2][128][40];   // [buf][m][k]
  __shared__ __align__(16) u16 Bs[2][32][136];   // [buf][k][n]

  const int tid  = threadIdx.x;
  const int lane = tid & 31;
  const int lh   = lane >> 4;     // 0/1 half-wave
  const int lm   = lane & 15;
  const int w    = tid >> 5;
  const int wm   = w & 3;         // wave M index 0..3
  const int wn   = w >> 2;        // wave N index 0..1
  const int rowBase = blockIdx.y * 128;
  const int colBase = blockIdx.x * 128;

  v8f vzero = {};
  v8f acc[2][4];
#pragma unroll
  for (int i = 0; i < 2; ++i)
#pragma unroll
    for (int j = 0; j < 4; ++j) acc[i][j] = vzero;

  // one WMMA K-step per stage: A frags 2x(b128 pair), B frags 4x u16-gather
  auto compute_tile = [&](int buf) {
    Frag16 af[2], bfr[4];
#pragma unroll
    for (int mi = 0; mi < 2; ++mi) {
      const int r   = wm * 32 + mi * 16 + lm;
      const int kb2 = lh * 8;
      af[mi].q[0] = *(const uint4*)&As[buf][r][kb2];       // K = kb2 + 0..7
      af[mi].q[1] = *(const uint4*)&As[buf][r][kb2 + 16];  // K = kb2 + 16..23
    }
#pragma unroll
    for (int ni = 0; ni < 4; ++ni) {
      const int n  = wn * 64 + ni * 16 + lm;
      const int kb = lh * 8;                               // K = kb + 0..7, kb + 16..23
#pragma unroll
      for (int e = 0; e < 16; ++e) bfr[ni].s[e] = Bs[buf][kb + e + (e >= 8 ? 8 : 0)][n];
    }
#pragma unroll
    for (int mi = 0; mi < 2; ++mi)
#pragma unroll
      for (int ni = 0; ni < 4; ++ni)
        acc[mi][ni] = __builtin_amdgcn_wmma_f32_16x16x32_bf16(
            false, af[mi].b, false, bfr[ni].b, (short)0, acc[mi][ni],
            false, false);
  };

#if HAS_TDM
  {
    // TDM issued once per wave; pick wave 0 with a scalar branch
    // (TDM ignores EXEC, so a lane-level guard would not stop other waves).
    const bool tw = (__builtin_amdgcn_readfirstlane(tid) >> 5) == 0;
    auto issue = [&](int buf, int k0) {
      tdm_load_2d_u16(lds_addr_of(&As[buf][0][0]),
                      A + (size_t)rowBase * K + k0,
                      /*tile*/ 32, 128, /*tensor*/ 32, 128,
                      (unsigned long long)K, /*pad 16DW+4DW*/ 3, 3);
      tdm_load_2d_u16(lds_addr_of(&Bs[buf][0][0]),
                      B + (size_t)k0 * N + colBase,
                      /*tile*/ 128, 32, /*tensor*/ 128, 32,
                      (unsigned long long)N, /*pad 64DW+4DW*/ 5, 3);
    };
    const int nk = K >> 5;
    if (tw) {
      issue(0, 0);
      __builtin_amdgcn_s_wait_tensorcnt(0);
    }
    __syncthreads();
    for (int s = 0; s < nk; ++s) {
      if (tw && (s + 1 < nk)) issue((s + 1) & 1, (s + 1) << 5);
      compute_tile(s & 1);
      if (tw) __builtin_amdgcn_s_wait_tensorcnt(0);
      __syncthreads();
    }
  }
#else
  {
    const int rowA = tid >> 1, colA = (tid & 1) * 16;   // 128x32 A tile
    const int rowB = tid >> 3, colB = (tid & 7) * 16;   // 32x128 B tile
    for (int k0 = 0; k0 < K; k0 += 32) {
      const u16* ga = A + (size_t)(rowBase + rowA) * K + k0 + colA;
      uint4 a0 = *(const uint4*)(ga);
      uint4 a1 = *(const uint4*)(ga + 8);
      const u16* gb = B + (size_t)(k0 + rowB) * N + colBase + colB;
      uint4 b0 = *(const uint4*)(gb);
      uint4 b1 = *(const uint4*)(gb + 8);
      if (k0 + 32 < K) {
        __builtin_prefetch(ga + 32, 0, 3);
        __builtin_prefetch(B + (size_t)(k0 + 32 + rowB) * N + colBase + colB,
                           0, 3);
      }
      __syncthreads();
      *(uint4*)&As[0][rowA][colA]     = a0;
      *(uint4*)&As[0][rowA][colA + 8] = a1;
      *(uint4*)&Bs[0][rowB][colB]     = b0;
      *(uint4*)&Bs[0][rowB][colB + 8] = b1;
      __syncthreads();
      compute_tile(0);
      __syncthreads();
    }
  }
#endif

  // C layout: element r of lane -> M = r + 8*(lane>>4), N = lane&15
#pragma unroll
  for (int mi = 0; mi < 2; ++mi)
#pragma unroll
    for (int ni = 0; ni < 4; ++ni)
#pragma unroll
      for (int r = 0; r < 8; ++r) {
        const int row = rowBase + wm * 32 + mi * 16 + r + 8 * lh;
        const int col = colBase + wn * 64 + ni * 16 + lm;
        float v = acc[mi][ni][r];
        if (EPI == 1) v = 0.5f * v * (1.0f + erff(v * 0.70710678118654752f));
        if (EPI == 2) {
          float* o = (float*)Cout;
          o[(size_t)row * N + col] = v + Res[(size_t)row * N + col];
        } else {
          u16* o = (u16*)Cout;
          o[(size_t)row * N + col] = f2bf(v);
        }
      }
}

// ---------------------------------------------------------------------------
// Flash attention. Block = (q-tile of 64, head, batch), 128 threads = 4 waves,
// each wave owns 16 query rows. K tile [key][dh] staged via TDM (stride 72
// u16 = 32DW rows + 4DW pad); V tile transposed [dh][key] by manual scatter;
// P bounced through per-wave LDS (C layout -> A layout) for the P@V WMMA.
// ---------------------------------------------------------------------------
__global__ __launch_bounds__(128) void attn_flash(
    const u16* __restrict__ Qb, const u16* __restrict__ Kb,
    const u16* __restrict__ Vb, u16* __restrict__ Yb) {
  __shared__ __align__(16) u16 Ks[64][72];      // [key][dh]
  __shared__ __align__(16) u16 Vt[64][72];      // [dh][key]
  __shared__ __align__(16) u16 Ps[4][16][72];   // per-wave P staging

  const int qt = blockIdx.x, h = blockIdx.y, b = blockIdx.z;
  const int tid = threadIdx.x, lane = tid & 31, w = tid >> 5;
  const int lh = lane >> 4, lm = lane & 15;
  const int q0   = qt * 64 + w * 16;            // this wave's query rows
  const int colh = h * DHEAD;
  const size_t rowOff = (size_t)b * T_SEQ;
#if HAS_TDM
  const bool tw = (__builtin_amdgcn_readfirstlane(tid) >> 5) == 0;
#endif

  // Q fragments (A layout), loaded straight from global
  Frag16 qf[2];
  {
    const u16* qrow = Qb + (rowOff + q0 + lm) * D_MODEL + colh;
#pragma unroll
    for (int i = 0; i < 2; ++i) {
      const int kb2 = i * 32 + lh * 16;
      qf[i].q[0] = *(const uint4*)(qrow + kb2);
      qf[i].q[1] = *(const uint4*)(qrow + kb2 + 8);
    }
  }

  float row_m[8], row_l[8];
  v8f vzero = {};
  v8f o[4];
#pragma unroll
  for (int r = 0; r < 8; ++r) { row_m[r] = -__builtin_inff(); row_l[r] = 0.f; }
#pragma unroll
  for (int j = 0; j < 4; ++j) o[j] = vzero;

  const int ldk = tid >> 1;          // key row this thread loads
  const int ldc = (tid & 1) * 32;    // dh column block

  for (int kt = 0; kt <= qt; ++kt) {             // causal key-tile loop
    {
      Frag32 vv;
      const u16* vr = Vb + (rowOff + kt * 64 + ldk) * D_MODEL + colh + ldc;
      vv.q[0] = *(const uint4*)(vr);
      vv.q[1] = *(const uint4*)(vr + 8);
      vv.q[2] = *(const uint4*)(vr + 16);
      vv.q[3] = *(const uint4*)(vr + 24);
#if !HAS_TDM
      const u16* kr = Kb + (rowOff + kt * 64 + ldk) * D_MODEL + colh + ldc;
      uint4 k0 = *(const uint4*)(kr);
      uint4 k1 = *(const uint4*)(kr + 8);
      uint4 k2 = *(const uint4*)(kr + 16);
      uint4 k3 = *(const uint4*)(kr + 24);
#endif
      __syncthreads();                           // previous tile consumed
#if HAS_TDM
      if (tw)                                     // DMA K tile during V scatter
        tdm_load_2d_u16(lds_addr_of(&Ks[0][0]),
                        Kb + (rowOff + kt * 64) * D_MODEL + colh,
                        /*tile*/ 64, 64, /*tensor*/ 64, 64,
                        (unsigned long long)D_MODEL, /*pad 32DW+4DW*/ 4, 3);
#else
      *(uint4*)&Ks[ldk][ldc]      = k0;
      *(uint4*)&Ks[ldk][ldc + 8]  = k1;
      *(uint4*)&Ks[ldk][ldc + 16] = k2;
      *(uint4*)&Ks[ldk][ldc + 24] = k3;
#endif
#pragma unroll
      for (int j = 0; j < 32; ++j) Vt[ldc + j][ldk] = vv.s[j];
#if HAS_TDM
      if (tw) __builtin_amdgcn_s_wait_tensorcnt(0);
#endif
      __syncthreads();
    }

    // S = Q K^T : B operand is K^T, i.e. Ks[key][dh] read as [N=key][K=dh]
    v8f s[4];
#pragma unroll
    for (int ni = 0; ni < 4; ++ni) {
      s[ni] = vzero;
#pragma unroll
      for (int i = 0; i < 2; ++i) {
        Frag16 bfr;
        const int kb = i * 32 + lh * 16;
        bfr.q[0] = *(const uint4*)&Ks[ni * 16 + lm][kb];
        bfr.q[1] = *(const uint4*)&Ks[ni * 16 + lm][kb + 8];
        s[ni] = __builtin_amdgcn_wmma_f32_16x16x32_bf16(
            false, qf[i].b, false, bfr.b, (short)0, s[ni], false, false);
      }
    }

    // online softmax on C layout: row = r + 8*lh, cols across 16-lane halves
#pragma unroll
    for (int r = 0; r < 8; ++r) {
      const int qpos = q0 + r + 8 * lh;
      float mx = row_m[r];
#pragma unroll
      for (int ni = 0; ni < 4; ++ni) {
        const int kpos = kt * 64 + ni * 16 + lm;
        float sv = s[ni][r] * 0.125f;            // 1/sqrt(64)
        sv = (kpos > qpos) ? -__builtin_inff() : sv;
        s[ni][r] = sv;
        mx = fmaxf(mx, sv);
      }
#pragma unroll
      for (int off = 1; off < 16; off <<= 1)
        mx = fmaxf(mx, __shfl_xor(mx, off, 32));
      const float alpha = __expf(row_m[r] - mx);
      float lsum = 0.f;
#pragma unroll
      for (int ni = 0; ni < 4; ++ni) {
        const float p = __expf(s[ni][r] - mx);
        s[ni][r] = p;
        lsum += p;
      }
#pragma unroll
      for (int off = 1; off < 16; off <<= 1)
        lsum += __shfl_xor(lsum, off, 32);
      row_l[r] = row_l[r] * alpha + lsum;
      row_m[r] = mx;
#pragma unroll
      for (int j = 0; j < 4; ++j) o[j][r] *= alpha;
    }

    // P: C layout -> per-wave LDS -> A layout (cross-lane dep within wave:
    // DS ops are wave-ordered; s_wait_dscnt 0 guards the readback).
#pragma unroll
    for (int ni = 0; ni < 4; ++ni)
#pragma unroll
      for (int r = 0; r < 8; ++r)
        Ps[w][r + 8 * lh][ni * 16 + lm] = f2bf(s[ni][r]);
    asm volatile("s_wait_dscnt 0" ::: "memory");

    // O += P @ V  (contraction over 64 keys = 2 WMMA K-steps)
#pragma unroll
    for (int i = 0; i < 2; ++i) {
      Frag16 pf;
      const int kb2 = i * 32 + lh * 16;
      pf.q[0] = *(const uint4*)&Ps[w][lm][kb2];
      pf.q[1] = *(const uint4*)&Ps[w][lm][kb2 + 8];
#pragma unroll
      for (int ni = 0; ni < 4; ++ni) {
        Frag16 vfr;
        const int kb = i * 32 + lh * 16;
        vfr.q[0] = *(const uint4*)&Vt[ni * 16 + lm][kb];
        vfr.q[1] = *(const uint4*)&Vt[ni * 16 + lm][kb + 8];
        o[ni] = __builtin_amdgcn_wmma_f32_16x16x32_bf16(
            false, pf.b, false, vfr.b, (short)0, o[ni], false, false);
      }
    }
  }

  // normalize and store y (bf16 for the Wo GEMM)
#pragma unroll
  for (int ni = 0; ni < 4; ++ni)
#pragma unroll
    for (int r = 0; r < 8; ++r) {
      const int row = q0 + r + 8 * lh;
      const int col = colh + ni * 16 + lm;
      Yb[(rowOff + row) * D_MODEL + col] = f2bf(o[ni][r] / row_l[r]);
    }
}

// ---------------------------------------------------------------------------
// Host-side orchestration
// ---------------------------------------------------------------------------
extern "C" void kernel_launch(void* const* d_in, const int* in_sizes, int n_in,
                              void* d_out, int out_size, void* d_ws,
                              size_t ws_size, hipStream_t stream) {
  const float* x  = (const float*)d_in[0];
  const float* Wq = (const float*)d_in[1];
  const float* Wk = (const float*)d_in[2];
  const float* Wv = (const float*)d_in[3];
  const float* Wo = (const float*)d_in[4];
  const float* W1 = (const float*)d_in[5];
  const float* W2 = (const float*)d_in[6];
  const float* g1 = (const float*)d_in[7];
  const float* g2 = (const float*)d_in[8];

  const int NT = BATCH * T_SEQ;                  // 8192 tokens

  char* ws = (char*)d_ws;
  size_t off = 0;
  auto alloc = [&](size_t bytes) -> void* {
    void* p = ws + off;
    off += (bytes + 255) & ~(size_t)255;
    return p;
  };
  u16* Wq_b = (u16*)alloc((size_t)D_MODEL * D_MODEL * 2);
  u16* Wk_b = (u16*)alloc((size_t)D_MODEL * D_MODEL * 2);
  u16* Wv_b = (u16*)alloc((size_t)D_MODEL * D_MODEL * 2);
  u16* Wo_b = (u16*)alloc((size_t)D_MODEL * D_MODEL * 2);
  u16* W1_b = (u16*)alloc((size_t)D_MODEL * DFF * 2);
  u16* W2_b = (u16*)alloc((size_t)DFF * D_MODEL * 2);
  u16* Xn   = (u16*)alloc((size_t)NT * D_MODEL * 2);
  u16* Qb   = (u16*)alloc((size_t)NT * D_MODEL * 2);
  u16* Kb   = (u16*)alloc((size_t)NT * D_MODEL * 2);
  u16* Vb   = (u16*)alloc((size_t)NT * D_MODEL * 2);
  u16* Yb   = (u16*)alloc((size_t)NT * D_MODEL * 2);
  float* X1 = (float*)alloc((size_t)NT * D_MODEL * 4);
  u16* Hb   = (u16*)alloc((size_t)NT * DFF * 2);

  // weight conversion (every call: launch must be stateless)
  cvt_f32_bf16<<<512, 256, 0, stream>>>(Wq, Wq_b, D_MODEL * D_MODEL);
  cvt_f32_bf16<<<512, 256, 0, stream>>>(Wk, Wk_b, D_MODEL * D_MODEL);
  cvt_f32_bf16<<<512, 256, 0, stream>>>(Wv, Wv_b, D_MODEL * D_MODEL);
  cvt_f32_bf16<<<512, 256, 0, stream>>>(Wo, Wo_b, D_MODEL * D_MODEL);
  cvt_f32_bf16<<<1024, 256, 0, stream>>>(W1, W1_b, D_MODEL * DFF);
  cvt_f32_bf16<<<1024, 256, 0, stream>>>(W2, W2_b, DFF * D_MODEL);

  // 1) xn = rmsnorm(x, g1)
  rmsnorm_bf16<<<NT, 256, 0, stream>>>(x, g1, Xn);

  // 2) Q/K/V projections
  dim3 gproj(D_MODEL / 128, NT / 128);           // (8, 64)
  gemm_bf16_128x128<0><<<gproj, 256, 0, stream>>>(Xn, Wq_b, Qb, nullptr,
                                                  NT, D_MODEL, D_MODEL);
  gemm_bf16_128x128<0><<<gproj, 256, 0, stream>>>(Xn, Wk_b, Kb, nullptr,
                                                  NT, D_MODEL, D_MODEL);
  gemm_bf16_128x128<0><<<gproj, 256, 0, stream>>>(Xn, Wv_b, Vb, nullptr,
                                                  NT, D_MODEL, D_MODEL);

  // 3) causal flash attention
  dim3 gattn(T_SEQ / 64, NHEADS, BATCH);         // (32, 16, 4)
  attn_flash<<<gattn, 128, 0, stream>>>(Qb, Kb, Vb, Yb);

  // 4) x1 = x + y @ Wo
  gemm_bf16_128x128<2><<<gproj, 256, 0, stream>>>(Yb, Wo_b, X1, x,
                                                  NT, D_MODEL, D_MODEL);

  // 5) xn = rmsnorm(x1, g2)
  rmsnorm_bf16<<<NT, 256, 0, stream>>>(X1, g2, Xn);

  // 6) h = gelu(xn @ W1)
  dim3 gff1(DFF / 128, NT / 128);                // (32, 64)
  gemm_bf16_128x128<1><<<gff1, 256, 0, stream>>>(Xn, W1_b, Hb, nullptr,
                                                 NT, DFF, D_MODEL);

  // 7) out = x1 + h @ W2
  gemm_bf16_128x128<2><<<gproj, 256, 0, stream>>>(Hb, W2_b, (float*)d_out, X1,
                                                  NT, D_MODEL, DFF);
}
